// MultiHeadAttention_42434276884695
// MI455X (gfx1250) — matrix operand order repaired and checked by reading, not compiled, not run on hardware
//
#include <hip/hip_runtime.h>
#include <hip/hip_bf16.h>
#include <math.h>

typedef __bf16 bf16_t;
typedef __attribute__((ext_vector_type(4)))  __bf16 v4bf;
typedef __attribute__((ext_vector_type(8)))  __bf16 v8bf;
typedef __attribute__((ext_vector_type(16))) __bf16 v16bf;
typedef __attribute__((ext_vector_type(8)))  float  v8f;

#define BATCH 2
#define CTX   2048
#define EMB   1024
#define NH    16
#define HD    64
#define MTOT  (BATCH * CTX)   // 4096 rows

#if defined(__gfx1250__)
#define USE_ASYNC_LDS 1
typedef __attribute__((address_space(3))) char las_char;
#else
#define USE_ASYNC_LDS 0
#endif

union ABf { v16bf v; v8bf h[2]; };

// ---- WMMA fragment loaders (contiguous k order, the same in A and B) ----

// A 16x32 bf16: lane provides row (row0 + lane%16); element e holds
// K = (lane/16)*16 + e  -> one contiguous 16-half (32B) run.
__device__ __forceinline__ v16bf load_a16(const bf16_t* __restrict__ base,
                                          int ldk, int row0, int k0, int lane) {
  const int g = lane >> 4;
  const bf16_t* p = base + (size_t)(row0 + (lane & 15)) * ldk + (k0 + g * 16);
  return *(const v16bf*)(p);
}

// B 32x16 bf16: lane provides col (col0 + lane%16); element e holds
// K = (lane/16)*16 + e  -> one contiguous 16-half (32B) run.
// base is row-major [col][k] (i.e. W[n][k] so B[k][n] = W[n][k]).
__device__ __forceinline__ v16bf load_b16(const bf16_t* __restrict__ base,
                                          int ldk, int col0, int k0, int lane) {
  const int g = lane >> 4;
  const bf16_t* p = base + (size_t)(col0 + (lane & 15)) * ldk + (k0 + g * 16);
  return *(const v16bf*)(p);
}

__device__ __forceinline__ v8f wmma_bf16(v16bf a, v16bf b, v8f c) {
  return __builtin_amdgcn_wmma_f32_16x16x32_bf16(
      /*neg_a=*/false, a, /*neg_b=*/false, b,
      /*c_mod=*/(short)0, c, /*reuse_a=*/false, /*reuse_b=*/false);
}

// ---------------- Kernel 0: f32 -> bf16 conversion ----------------
__global__ void __launch_bounds__(256) cvt_f32_bf16(const float* __restrict__ in,
                                                    bf16_t* __restrict__ out, int n4) {
  int i = blockIdx.x * blockDim.x + threadIdx.x;
  if (i < n4) {
    float4 f = ((const float4*)in)[i];
    v4bf o;
    o.x = (bf16_t)f.x; o.y = (bf16_t)f.y; o.z = (bf16_t)f.z; o.w = (bf16_t)f.w;
    ((v4bf*)out)[i] = o;
  }
}

// ---------------- Kernel 1: QKV projection + RoPE ----------------
// Each wave: 32(M) x 64(N) tile of Y = xb @ W^T. N-span == one head.
__global__ void __launch_bounds__(128) qkv_rope_kernel(
    const bf16_t* __restrict__ xb,  const bf16_t* __restrict__ Wqb,
    const bf16_t* __restrict__ Wkb, const bf16_t* __restrict__ Wvb,
    bf16_t* __restrict__ Qb, bf16_t* __restrict__ Kb, bf16_t* __restrict__ Vt) {
  const int lane = threadIdx.x & 31;
  const int wid  = blockIdx.x * 4 + (threadIdx.x >> 5);
  const int TPM  = (MTOT / 32) * (EMB / 64);   // 2048 tiles per matrix
  const int mat  = wid / TPM;                  // 0=Q 1=K 2=V
  const int rem  = wid % TPM;
  const int tm   = rem >> 4;
  const int tn   = rem & 15;
  const int row0 = tm * 32, col0 = tn * 64;
  const bf16_t* W = (mat == 0) ? Wqb : (mat == 1) ? Wkb : Wvb;

  const v8f vzero = {0.f,0.f,0.f,0.f,0.f,0.f,0.f,0.f};
  v8f acc[2][4];
  #pragma unroll
  for (int i = 0; i < 2; ++i)
    #pragma unroll
    for (int j = 0; j < 4; ++j) acc[i][j] = vzero;

  for (int k0 = 0; k0 < EMB; k0 += 32) {
    v16bf a0 = load_a16(xb, EMB, row0,      k0, lane);
    v16bf a1 = load_a16(xb, EMB, row0 + 16, k0, lane);
    #pragma unroll
    for (int j = 0; j < 4; ++j) {
      v16bf b = load_b16(W, EMB, col0 + j * 16, k0, lane);
      acc[0][j] = wmma_bf16(a0, b, acc[0][j]);
      acc[1][j] = wmma_bf16(a1, b, acc[1][j]);
    }
  }

  const int g = lane >> 4, n = lane & 15;
  const int h = tn;  // head index (64-wide span == head)
  if (mat < 2) {
    // rotate-half RoPE: pair column d (=j*16+n, j<2) with d+32 (tile j+2)
    bf16_t* dst = (mat == 0) ? Qb : Kb;
    #pragma unroll
    for (int j = 0; j < 2; ++j) {
      const float invf = __expf(-(float)(j * 16 + n) * (9.210340371976184f / 32.0f));
      #pragma unroll
      for (int i = 0; i < 2; ++i) {
        #pragma unroll
        for (int r = 0; r < 8; ++r) {
          const int row = row0 + i * 16 + 8 * g + r;
          const int c   = row & (CTX - 1);
          const int bb  = row >> 11;
          const float ang = (float)c * invf;
          const float sn = sinf(ang), cs = cosf(ang);
          const float v1 = acc[i][j][r], v2 = acc[i][j + 2][r];
          bf16_t* q = dst + (((size_t)bb * NH + h) * CTX + c) * HD;
          q[j * 16 + n]      = (bf16_t)(v1 * cs - v2 * sn);
          q[j * 16 + n + 32] = (bf16_t)(v2 * cs + v1 * sn);
        }
      }
    }
  } else {
    // V stored transposed: Vt[b][h][d][c] so P@V B-fragments are contiguous
    #pragma unroll
    for (int j = 0; j < 4; ++j)
      #pragma unroll
      for (int i = 0; i < 2; ++i)
        #pragma unroll
        for (int r = 0; r < 8; ++r) {
          const int row = row0 + i * 16 + 8 * g + r;
          const int c   = row & (CTX - 1);
          const int bb  = row >> 11;
          const int d   = j * 16 + n;
          Vt[(((size_t)bb * NH + h) * HD + d) * CTX + c] = (bf16_t)acc[i][j][r];
        }
  }
}

// ---------------- Kernel 2: causal flash attention ----------------
// Per wave: 32 query rows of one (b,h); online softmax over 64-key blocks.
// K block (contiguous 8KB in [c][d] layout) is staged into LDS with
// gfx1250 global_load_async_to_lds_b128 (ASYNCcnt-tracked, issued via asm).
__global__ void __launch_bounds__(128) attn_kernel(
    const bf16_t* __restrict__ Qb, const bf16_t* __restrict__ Kb,
    const bf16_t* __restrict__ Vt, bf16_t* __restrict__ Ab) {
  __shared__ __align__(32) bf16_t pbuf[4][32 * 64];    // wave-private P staging
#if USE_ASYNC_LDS
  __shared__ __align__(32) bf16_t kstage[4][64 * 64];  // wave-private K staging
#endif
  const int lane  = threadIdx.x & 31;
  const int wslot = threadIdx.x >> 5;
  const int wid   = blockIdx.x * 4 + wslot;
  const int bh    = wid >> 6;        // (b*16 + h)
  const int qt    = wid & 63;
  const int q0    = qt * 32;
  const bf16_t* Qh = Qb + (size_t)bh * CTX * HD;
  const bf16_t* Kh = Kb + (size_t)bh * CTX * HD;
  const bf16_t* Vh = Vt + (size_t)bh * HD * CTX;
  bf16_t* pw = pbuf[wslot];
#if USE_ASYNC_LDS
  bf16_t* kw = kstage[wslot];
#endif
  const int g = lane >> 4, n = lane & 15;

  v16bf aq[2][2];
  #pragma unroll
  for (int i = 0; i < 2; ++i)
    #pragma unroll
    for (int kk = 0; kk < 2; ++kk)
      aq[i][kk] = load_a16(Qh, HD, q0 + i * 16, kk * 32, lane);

  const v8f vzero = {0.f,0.f,0.f,0.f,0.f,0.f,0.f,0.f};
  float mrow[2][8], lrow[2][8];
  v8f accv[2][4];
  #pragma unroll
  for (int i = 0; i < 2; ++i) {
    #pragma unroll
    for (int r = 0; r < 8; ++r) { mrow[i][r] = -INFINITY; lrow[i][r] = 0.f; }
    #pragma unroll
    for (int j = 0; j < 4; ++j) accv[i][j] = vzero;
  }

  const int lastb = (q0 + 31) >> 6;
  for (int kb = 0; kb <= lastb; ++kb) {
    const int kbase = kb * 64;
    if (kb < lastb) {  // gfx1250 global_prefetch_b8 for next K/V blocks
      __builtin_prefetch(Kh + (size_t)(kbase + 64 + lane) * HD, 0, 0);
      __builtin_prefetch(Vh + (size_t)lane * CTX + kbase + 64, 0, 0);
    }
#if USE_ASYNC_LDS
    {
      // WAR guard: previous iteration's ds_load fragments must be in VGPRs
      // before the async engine overwrites the staging buffer.
      asm volatile("s_wait_dscnt 0x0" ::: "memory");
      const unsigned lds0 =
          (unsigned)(size_t)(las_char*)((char*)kw + lane * 16);
      const unsigned long long ga0 =
          (unsigned long long)(size_t)((const char*)(Kh + (size_t)kbase * HD) +
                                       lane * 16);
      #pragma unroll
      for (int t = 0; t < 16; ++t)  // 16 x 32 lanes x 16B = 8KB K block
        asm volatile("global_load_async_to_lds_b128 %0, %1, off"
                     :: "v"(lds0 + (unsigned)(t * 512)),
                        "v"(ga0 + (unsigned long long)(t * 512))
                     : "memory");
      // async LDS writes are tracked by ASYNCcnt, not DScnt
      asm volatile("s_wait_asynccnt 0x0" ::: "memory");
    }
#endif
    // S = Q @ K^T (K rows [c][d] are exactly the B layout, contiguous in d)
    v8f s[2][4];
    #pragma unroll
    for (int j = 0; j < 4; ++j) {
#if USE_ASYNC_LDS
      v16bf bk0 = load_b16(kw, HD, j * 16, 0,  lane);
      v16bf bk1 = load_b16(kw, HD, j * 16, 32, lane);
#else
      v16bf bk0 = load_b16(Kh, HD, kbase + j * 16, 0,  lane);
      v16bf bk1 = load_b16(Kh, HD, kbase + j * 16, 32, lane);
#endif
      #pragma unroll
      for (int i = 0; i < 2; ++i) {
        v8f t = wmma_bf16(aq[i][0], bk0, vzero);
        t     = wmma_bf16(aq[i][1], bk1, t);
        s[i][j] = t;
      }
    }
    const bool diag = (kb == lastb);
    #pragma unroll
    for (int i = 0; i < 2; ++i)
      #pragma unroll
      for (int j = 0; j < 4; ++j)
        #pragma unroll
        for (int r = 0; r < 8; ++r) {
          float v = s[i][j][r] * 0.125f;  // 1/sqrt(64)
          if (diag) {
            const int key = kbase + j * 16 + n;
            const int row = q0 + i * 16 + 8 * g + r;
            if (key > row) v = -INFINITY;
          }
          s[i][j][r] = v;
        }
    // online softmax per row (row = r + 8g within each 16-row tile)
    #pragma unroll
    for (int i = 0; i < 2; ++i) {
      #pragma unroll
      for (int r = 0; r < 8; ++r) {
        float pm = s[i][0][r];
        #pragma unroll
        for (int j = 1; j < 4; ++j) pm = fmaxf(pm, s[i][j][r]);
        #pragma unroll
        for (int off = 1; off < 16; off <<= 1)
          pm = fmaxf(pm, __shfl_xor(pm, off, 32));
        const float mnew  = fmaxf(mrow[i][r], pm);
        const float alpha = __expf(mrow[i][r] - mnew);
        float ps = 0.f;
        #pragma unroll
        for (int j = 0; j < 4; ++j) {
          const float p = __expf(s[i][j][r] - mnew);
          s[i][j][r] = p;
          ps += p;
        }
        #pragma unroll
        for (int off = 1; off < 16; off <<= 1)
          ps += __shfl_xor(ps, off, 32);
        lrow[i][r] = lrow[i][r] * alpha + ps;
        mrow[i][r] = mnew;
        #pragma unroll
        for (int j = 0; j < 4; ++j) accv[i][j][r] *= alpha;
        #pragma unroll
        for (int j = 0; j < 4; ++j)
          pw[(i * 16 + 8 * g + r) * 64 + j * 16 + n] = (bf16_t)s[i][j][r];
      }
    }
    // O += P @ V : P re-read from LDS in A layout, V from transposed Vt
    v16bf ap[2][2];
    #pragma unroll
    for (int i = 0; i < 2; ++i)
      #pragma unroll
      for (int kk = 0; kk < 2; ++kk)
        ap[i][kk] = load_a16(pw, 64, i * 16, kk * 32, lane);
    #pragma unroll
    for (int dj = 0; dj < 4; ++dj) {
      v16bf bv0 = load_b16(Vh, CTX, dj * 16, kbase,      lane);
      v16bf bv1 = load_b16(Vh, CTX, dj * 16, kbase + 32, lane);
      #pragma unroll
      for (int i = 0; i < 2; ++i) {
        accv[i][dj] = wmma_bf16(ap[i][0], bv0, accv[i][dj]);
        accv[i][dj] = wmma_bf16(ap[i][1], bv1, accv[i][dj]);
      }
    }
  }
  // epilogue: divide by row sum, write hidden[b, c, h*64+d] as bf16
  const int b = bh >> 4, h = bh & 15;
  #pragma unroll
  for (int i = 0; i < 2; ++i)
    #pragma unroll
    for (int dj = 0; dj < 4; ++dj)
      #pragma unroll
      for (int r = 0; r < 8; ++r) {
        const int row = q0 + i * 16 + 8 * g + r;
        const float o = accv[i][dj][r] / lrow[i][r];
        Ab[((size_t)b * CTX + row) * EMB + h * HD + dj * 16 + n] = (bf16_t)o;
      }
}

// ---------------- Kernel 3: output projection (f32 out) ----------------
__global__ void __launch_bounds__(128) oproj_kernel(
    const bf16_t* __restrict__ Ab, const bf16_t* __restrict__ Wob,
    float* __restrict__ out) {
  const int lane = threadIdx.x & 31;
  const int wid  = blockIdx.x * 4 + (threadIdx.x >> 5);
  const int tm   = wid >> 4, tn = wid & 15;
  const int row0 = tm * 32, col0 = tn * 64;

  const v8f vzero = {0.f,0.f,0.f,0.f,0.f,0.f,0.f,0.f};
  v8f acc[2][4];
  #pragma unroll
  for (int i = 0; i < 2; ++i)
    #pragma unroll
    for (int j = 0; j < 4; ++j) acc[i][j] = vzero;

  for (int k0 = 0; k0 < EMB; k0 += 32) {
    v16bf a0 = load_a16(Ab, EMB, row0,      k0, lane);
    v16bf a1 = load_a16(Ab, EMB, row0 + 16, k0, lane);
    #pragma unroll
    for (int j = 0; j < 4; ++j) {
      v16bf b = load_b16(Wob, EMB, col0 + j * 16, k0, lane);
      acc[0][j] = wmma_bf16(a0, b, acc[0][j]);
      acc[1][j] = wmma_bf16(a1, b, acc[1][j]);
    }
  }
  const int g = lane >> 4, n = lane & 15;
  #pragma unroll
  for (int i = 0; i < 2; ++i)
    #pragma unroll
    for (int j = 0; j < 4; ++j)
      #pragma unroll
      for (int r = 0; r < 8; ++r)
        out[(size_t)(row0 + i * 16 + 8 * g + r) * EMB + col0 + j * 16 + n] =
            acc[i][j][r];
}

// ---------------- launch ----------------
extern "C" void kernel_launch(void* const* d_in, const int* in_sizes, int n_in,
                              void* d_out, int out_size, void* d_ws, size_t ws_size,
                              hipStream_t stream) {
  (void)in_sizes; (void)n_in; (void)out_size; (void)ws_size;
  const float* x  = (const float*)d_in[0];
  const float* Wq = (const float*)d_in[1];
  const float* Wk = (const float*)d_in[2];
  const float* Wv = (const float*)d_in[3];
  const float* Wo = (const float*)d_in[4];

  char* ws = (char*)d_ws;
  const size_t MB = 1024 * 1024;
  bf16_t* xb  = (bf16_t*)(ws + 0 * MB);   // 8 MB  [4096,1024]
  bf16_t* Wqb = (bf16_t*)(ws + 8 * MB);   // 2 MB
  bf16_t* Wkb = (bf16_t*)(ws + 10 * MB);  // 2 MB
  bf16_t* Wvb = (bf16_t*)(ws + 12 * MB);  // 2 MB
  bf16_t* Wob = (bf16_t*)(ws + 14 * MB);  // 2 MB
  bf16_t* Qb  = (bf16_t*)(ws + 16 * MB);  // 8 MB  [b,h,c,d]
  bf16_t* Kb  = (bf16_t*)(ws + 24 * MB);  // 8 MB  [b,h,c,d]
  bf16_t* Vt  = (bf16_t*)(ws + 32 * MB);  // 8 MB  [b,h,d,c]
  bf16_t* Ab  = (bf16_t*)(ws + 40 * MB);  // 8 MB  [b*c, h*64+d]

  const int NX = MTOT * EMB / 4;  // float4 count
  const int NW = EMB * EMB / 4;
  cvt_f32_bf16<<<NX / 256, 256, 0, stream>>>(x,  xb,  NX);
  cvt_f32_bf16<<<NW / 256, 256, 0, stream>>>(Wq, Wqb, NW);
  cvt_f32_bf16<<<NW / 256, 256, 0, stream>>>(Wk, Wkb, NW);
  cvt_f32_bf16<<<NW / 256, 256, 0, stream>>>(Wv, Wvb, NW);
  cvt_f32_bf16<<<NW / 256, 256, 0, stream>>>(Wo, Wob, NW);

  qkv_rope_kernel<<<1536, 128, 0, stream>>>(xb, Wqb, Wkb, Wvb, Qb, Kb, Vt);
  attn_kernel<<<512, 128, 0, stream>>>(Qb, Kb, Vt, Ab);
  oproj_kernel<<<512, 128, 0, stream>>>(Ab, Wob, (float*)d_out);
}
